// GraphConsis_27977416966331
// MI455X (gfx1250) — matrix operand order repaired and checked by reading, not compiled, not run on hardware
//
#include <hip/hip_runtime.h>

// ---------- types for WMMA ----------
typedef __attribute__((ext_vector_type(16))) __bf16          v16bf;
typedef __attribute__((ext_vector_type(16))) unsigned short  v16us;
typedef __attribute__((ext_vector_type(8)))  unsigned short  v8us;
typedef __attribute__((ext_vector_type(8)))  float           v8f;

// native fp32 -> bf16 (RNE) ; backend emits v_cvt_pk_bf16_f32
__device__ __forceinline__ unsigned pack2bf(float lo, float hi) {
  unsigned short l = __builtin_bit_cast(unsigned short, (__bf16)lo);
  unsigned short h = __builtin_bit_cast(unsigned short, (__bf16)hi);
  return (unsigned)l | ((unsigned)h << 16);
}

#define KSTEP  32   // K per WMMA step (bf16 WMMA K)
#define BN     64   // N columns per block (4 fragments of 16)
#define KP     40   // padded K stride in LDS (byte stride 80 -> 16B aligned frag reads)
#define NWAVES 8    // waves per 256-thread block; block tile = 128(M) x 64(N)

// C[M,N] = A[M,K] @ Bgathered[K,N]; fp32 in memory, bf16 WMMA, fp32 accumulate.
// GM: 0 -> row = k ; 1 -> row = idx1[k] ; 2 -> row = idx2[idx1[k]]
// Requires K % 64 == 0 (true for all call sites: 16384, 512, 4096, 256).
template <int GM>
__global__ __launch_bounds__(256)
void gemm_bf16_wmma(const float* __restrict__ A, long long sArel, int lda,
                    const float* __restrict__ Bm, long long sBrel, int ldb,
                    const int* __restrict__ idx1, long long sI1,
                    const int* __restrict__ idx2, long long sI2,
                    float* __restrict__ C, long long sCrel, int ldc,
                    int K)
{
  const int r = blockIdx.z;
  A  += (long long)r * sArel;
  Bm += (long long)r * sBrel;
  C  += (long long)r * sCrel;
  if (GM >= 1) idx1 += (long long)r * sI1;
  if (GM >= 2) idx2 += (long long)r * sI2;

  __shared__ alignas(16) unsigned short Blds[2][BN * KP];  // double-buffered col-major bf16 tile

  const int tid    = threadIdx.x;
  const int wave   = tid >> 5;
  const int lane   = tid & 31;
  const int laneLo = lane & 15;
  const int hiHalf = lane >> 4;
  const int kk     = (tid & 15) * 2;   // two consecutive B-tile rows staged by this thread
  const int c0     = (tid >> 4) * 4;   // 4-column chunk staged by this thread

  const int m0 = blockIdx.x * (NWAVES * 16) + wave * 16;
  const int n0 = blockIdx.y * BN;

  v8f acc[4] = {};

  const float* arow = A + (long long)(m0 + laneLo) * lda + hiHalf * 8;

  auto rowOf = [&](int k) -> long long {
    if (GM == 0) return k;
    int t = idx1[k];
    if (GM == 1) return t;
    return idx2[t];
  };

  // fetch tile k into registers: 2 B rows x 4 cols + 16 A values
  auto fetchTile = [&](int k, float4& b0, float4& b1,
                       float4& a0, float4& a1, float4& a2, float4& a3) {
    b0 = *(const float4*)(Bm + rowOf(k + kk) * ldb + (n0 + c0));
    b1 = *(const float4*)(Bm + rowOf(k + kk + 1) * ldb + (n0 + c0));
    a0 = *(const float4*)(arow + k);
    a1 = *(const float4*)(arow + k + 4);
    a2 = *(const float4*)(arow + k + 16);
    a3 = *(const float4*)(arow + k + 20);
  };

  // stage two rows x 4 cols of B (fp32 regs) into an LDS buffer as packed bf16
  auto stageB = [&](int buf, const float4& r0, const float4& r1) {
    unsigned short* base = &Blds[buf][0];
    *(unsigned*)(base + (c0 + 0) * KP + kk) = pack2bf(r0.x, r1.x);
    *(unsigned*)(base + (c0 + 1) * KP + kk) = pack2bf(r0.y, r1.y);
    *(unsigned*)(base + (c0 + 2) * KP + kk) = pack2bf(r0.z, r1.z);
    *(unsigned*)(base + (c0 + 3) * KP + kk) = pack2bf(r0.w, r1.w);
  };

  auto mkA = [&](const float4& a0, const float4& a1,
                 const float4& a2, const float4& a3) -> v16bf {
    v16bf af;
    af[0] =(__bf16)a0.x; af[1] =(__bf16)a0.y; af[2] =(__bf16)a0.z; af[3] =(__bf16)a0.w;
    af[4] =(__bf16)a1.x; af[5] =(__bf16)a1.y; af[6] =(__bf16)a1.z; af[7] =(__bf16)a1.w;
    af[8] =(__bf16)a2.x; af[9] =(__bf16)a2.y; af[10]=(__bf16)a2.z; af[11]=(__bf16)a2.w;
    af[12]=(__bf16)a3.x; af[13]=(__bf16)a3.y; af[14]=(__bf16)a3.z; af[15]=(__bf16)a3.w;
    return af;
  };

  auto loadB = [&](int buf, int f) -> v16bf {
    const unsigned short* bp = &Blds[buf][(f * 16 + laneLo) * KP + hiHalf * 8];
    v8us blo = *(const v8us*)bp;        // K = hiHalf*8 + 0..7
    v8us bhi = *(const v8us*)(bp + 16); // K = 16 + hiHalf*8 + 0..7
    v16us bu;
#pragma unroll
    for (int i = 0; i < 8; ++i) { bu[i] = blo[i]; bu[8 + i] = bhi[i]; }
    return __builtin_bit_cast(v16bf, bu);
  };

  // 4 B-fragment loads interleaved with 4 WMMAs against LDS buffer `buf`
  auto compute = [&](int buf, v16bf afrag) {
    v16bf bf0 = loadB(buf, 0);
    v16bf bf1 = loadB(buf, 1);
    acc[0] = __builtin_amdgcn_wmma_f32_16x16x32_bf16(false, afrag, false, bf0, (short)0, acc[0], false, false);
    v16bf bf2 = loadB(buf, 2);
    acc[1] = __builtin_amdgcn_wmma_f32_16x16x32_bf16(false, afrag, false, bf1, (short)0, acc[1], false, false);
    v16bf bf3 = loadB(buf, 3);
    acc[2] = __builtin_amdgcn_wmma_f32_16x16x32_bf16(false, afrag, false, bf2, (short)0, acc[2], false, false);
    acc[3] = __builtin_amdgcn_wmma_f32_16x16x32_bf16(false, afrag, false, bf3, (short)0, acc[3], false, false);
  };

  // ---- prologue: tile 0 -> register set X -> LDS buffer 0 ----
  float4 xb0, xb1, xa0, xa1, xa2, xa3;
  float4 yb0, yb1, ya0, ya1, ya2, ya3;
  fetchTile(0, xb0, xb1, xa0, xa1, xa2, xa3);
  stageB(0, xb0, xb1);
  __syncthreads();

  // ---- 2x-unrolled pipeline: X/buf0 and Y/buf1 alternate structurally ----
  for (int k0 = 0; k0 < K; k0 += 2 * KSTEP) {
    // half 1: compute tile k0 from buf0; prefetch+stage tile k0+32 into buf1
    fetchTile(k0 + KSTEP, yb0, yb1, ya0, ya1, ya2, ya3);
    compute(0, mkA(xa0, xa1, xa2, xa3));
    stageB(1, yb0, yb1);
    __syncthreads();

    // half 2: compute tile k0+32 from buf1; prefetch+stage tile k0+64 into buf0
    const bool hasNext = (k0 + 2 * KSTEP) < K;
    if (hasNext) fetchTile(k0 + 2 * KSTEP, xb0, xb1, xa0, xa1, xa2, xa3);
    compute(1, mkA(ya0, ya1, ya2, ya3));
    if (hasNext) stageB(0, xb0, xb1);
    __syncthreads();
  }

  // ---- store C: VGPR v -> (M = v + hiHalf*8, N = laneLo + f*16) ----
#pragma unroll
  for (int f = 0; f < 4; ++f) {
    float* crow = C + (long long)(m0 + hiHalf * 8) * ldc + (n0 + f * 16 + laneLo);
#pragma unroll
    for (int v = 0; v < 8; ++v)
      crow[(long long)v * ldc] = acc[f][v];
  }
}

// out[d, 0:Ka] = A[d, :];  out[d, Ka:Ka+Kb] = B[gather(d), :]
template <int GM>
__global__ __launch_bounds__(256)
void cat_gather(const float* __restrict__ A, long long sArel, int Ka,
                const float* __restrict__ Bm, long long sBrel, int Kb,
                const int* __restrict__ idx1, long long sI1,
                const int* __restrict__ idx2, long long sI2,
                float* __restrict__ out, long long sOrel)
{
  const int r = blockIdx.z;
  A   += (long long)r * sArel;
  Bm  += (long long)r * sBrel;
  out += (long long)r * sOrel;
  if (GM >= 1) idx1 += (long long)r * sI1;
  if (GM >= 2) idx2 += (long long)r * sI2;

  const int d = blockIdx.x;
  long long rowb = d;
  if (GM >= 1) rowb = idx1[d];
  if (GM >= 2) rowb = idx2[rowb];

  const float* arow = A  + (long long)d * Ka;
  const float* brow = Bm + rowb * Kb;
  float* orow = out + (long long)d * (Ka + Kb);
  for (int j = threadIdx.x; j < Ka; j += blockDim.x) orow[j]      = arow[j];
  for (int j = threadIdx.x; j < Kb; j += blockDim.x) orow[Ka + j] = brow[j];
}

// score = H @ att[0:Hd] + rel . att[Hd:2Hd];  att = softmax(score over D);  X = H * att[:,None]
__global__ __launch_bounds__(256)
void attention_softmax(const float* __restrict__ Hm, long long sHrel,
                       const float* __restrict__ relv,
                       const float* __restrict__ attv,
                       float* __restrict__ X, long long sXrel,
                       int D, int Hd)
{
  const int r = blockIdx.x;
  Hm += (long long)r * sHrel;
  X  += (long long)r * sXrel;
  const float* rv = relv + (long long)r * Hd;

  __shared__ float sc[4096];
  __shared__ float red[256];
  const int tid = threadIdx.x;
  const int H4 = Hd >> 2;

  float crel = 0.f;
  {
    const float4* r4 = (const float4*)rv;
    const float4* a4 = (const float4*)(attv + Hd);
    for (int j = 0; j < H4; ++j) {
      float4 rj = r4[j], aj = a4[j];
      crel += rj.x * aj.x + rj.y * aj.y + rj.z * aj.z + rj.w * aj.w;
    }
  }

  const float4* av4 = (const float4*)attv;
  for (int d = tid; d < D; d += 256) {
    const float4* h4 = (const float4*)(Hm + (long long)d * Hd);
    float s = crel;
    for (int j = 0; j < H4; ++j) {
      float4 hj = h4[j], aj = av4[j];
      s += hj.x * aj.x + hj.y * aj.y + hj.z * aj.z + hj.w * aj.w;
    }
    sc[d] = s;
  }
  __syncthreads();

  float m = -3.402823466e+38f;
  for (int d = tid; d < D; d += 256) m = fmaxf(m, sc[d]);
  red[tid] = m; __syncthreads();
  for (int off = 128; off > 0; off >>= 1) {
    if (tid < off) red[tid] = fmaxf(red[tid], red[tid + off]);
    __syncthreads();
  }
  const float mx = red[0];
  __syncthreads();

  float sum = 0.f;
  for (int d = tid; d < D; d += 256) {
    float e = __expf(sc[d] - mx);
    sc[d] = e; sum += e;
  }
  red[tid] = sum; __syncthreads();
  for (int off = 128; off > 0; off >>= 1) {
    if (tid < off) red[tid] += red[tid + off];
    __syncthreads();
  }
  const float inv = 1.f / red[0];

  for (int d = tid; d < D; d += 256) {
    const float w = sc[d] * inv;
    const float4* hi = (const float4*)(Hm + (long long)d * Hd);
    float4* xo = (float4*)(X + (long long)d * Hd);
    for (int j = 0; j < H4; ++j) {
      float4 h = hi[j];
      xo[j] = make_float4(h.x * w, h.y * w, h.z * w, h.w * w);
    }
  }
}

// sum over relations, row-L2-normalize, @ classifier_w, row softmax
__global__ __launch_bounds__(128)
void finalize(const float* __restrict__ X2, long long sXrel,
              const float* __restrict__ CW,
              float* __restrict__ out, int Hd, int Cd)
{
  const int row = blockIdx.x;
  const int j = threadIdx.x;
  float s = 0.f;
  for (int rr = 0; rr < 3; ++rr)
    s += X2[(long long)rr * sXrel + (long long)row * Hd + j];

  __shared__ float red[128];
  __shared__ float lg[2];
  red[j] = s * s; __syncthreads();
  for (int off = 64; off > 0; off >>= 1) {
    if (j < off) red[j] += red[j + off];
    __syncthreads();
  }
  const float inv = rsqrtf(fmaxf(red[0], 1e-12f));
  const float v = s * inv;

  for (int c = 0; c < Cd; ++c) {
    __syncthreads();
    red[j] = v * CW[(long long)j * Cd + c];
    __syncthreads();
    for (int off = 64; off > 0; off >>= 1) {
      if (j < off) red[j] += red[j + off];
      __syncthreads();
    }
    if (j == 0) lg[c] = red[0];
  }
  __syncthreads();
  if (j < Cd) {
    float mx = fmaxf(lg[0], lg[1]);
    float e0 = __expf(lg[0] - mx), e1 = __expf(lg[1] - mx);
    float denom = e0 + e1;
    out[(long long)row * Cd + j] = (j == 0 ? e0 : e1) / denom;
  }
}

extern "C" void kernel_launch(void* const* d_in, const int* in_sizes, int n_in,
                              void* d_out, int out_size, void* d_ws, size_t ws_size,
                              hipStream_t stream) {
  (void)in_sizes; (void)n_in; (void)out_size; (void)ws_size;
  const int R = 3, F = 256, H = 128, Cd = 2;
  const int N0 = 16384, S1 = 16384, D1 = 4096, S2 = 4096, D2 = 512;

  const float* features = (const float*)d_in[0];
  const int*   src      = (const int*)  d_in[1];   // [R,N0]
  const int*   s1       = (const int*)  d_in[2];   // [R,S1]
  const int*   d1       = (const int*)  d_in[3];   // [R,D1]
  const float* dif1     = (const float*)d_in[4];   // [R,D1,S1]
  const int*   s2       = (const int*)  d_in[5];   // [R,S2]
  const int*   d2       = (const int*)  d_in[6];   // [R,D2]
  const float* dif2     = (const float*)d_in[7];   // [R,D2,S2]
  const float* w1       = (const float*)d_in[8];   // [2F,H]
  const float* w2       = (const float*)d_in[9];   // [2H,H]
  const float* relv     = (const float*)d_in[10];  // [R,H]
  const float* attv     = (const float*)d_in[11];  // [2H,1]
  const float* cw       = (const float*)d_in[12];  // [H,Cd]
  float* out = (float*)d_out;

  float* ws   = (float*)d_ws;
  float* agg1 = ws;                                    // R*D1*F
  float* cat1 = agg1 + (size_t)R * D1 * F;             // R*D1*2F
  float* h1   = cat1 + (size_t)R * D1 * 2 * F;         // R*D1*H
  float* x1   = h1   + (size_t)R * D1 * H;             // R*D1*H
  float* agg2 = x1   + (size_t)R * D1 * H;             // R*D2*H
  float* cat2 = agg2 + (size_t)R * D2 * H;             // R*D2*2H
  float* h2   = cat2 + (size_t)R * D2 * 2 * H;         // R*D2*H
  float* x2   = h2   + (size_t)R * D2 * H;             // R*D2*H

  dim3 blk(256);

  // 1) agg1[r] = dif1[r] @ features[src[s1]]   (M=4096,N=256,K=16384)
  gemm_bf16_wmma<2><<<dim3(D1 / 128, F / BN, R), blk, 0, stream>>>(
      dif1, (long long)D1 * S1, S1,
      features, 0, F,
      s1, S1, src, N0,
      agg1, (long long)D1 * F, F,
      S1);

  // 2) cat1[r] = [agg1[r] | features[src[d1]]]
  cat_gather<2><<<dim3(D1, 1, R), blk, 0, stream>>>(
      agg1, (long long)D1 * F, F,
      features, 0, F,
      d1, D1, src, N0,
      cat1, (long long)D1 * 2 * F);

  // 3) h1[r] = cat1[r] @ w1                    (M=4096,N=128,K=512)
  gemm_bf16_wmma<0><<<dim3(D1 / 128, H / BN, R), blk, 0, stream>>>(
      cat1, (long long)D1 * 2 * F, 2 * F,
      w1, 0, H,
      nullptr, 0, nullptr, 0,
      h1, (long long)D1 * H, H,
      2 * F);

  // 4) x1[r] = h1[r] * softmax(attention)
  attention_softmax<<<dim3(R), blk, 0, stream>>>(
      h1, (long long)D1 * H, relv, attv, x1, (long long)D1 * H, D1, H);

  // 5) agg2[r] = dif2[r] @ x1[r][s2]           (M=512,N=128,K=4096)
  gemm_bf16_wmma<1><<<dim3(D2 / 128, H / BN, R), blk, 0, stream>>>(
      dif2, (long long)D2 * S2, S2,
      x1, (long long)D1 * H, H,
      s2, S2, nullptr, 0,
      agg2, (long long)D2 * H, H,
      S2);

  // 6) cat2[r] = [agg2[r] | x1[r][d2]]
  cat_gather<1><<<dim3(D2, 1, R), blk, 0, stream>>>(
      agg2, (long long)D2 * H, H,
      x1, (long long)D1 * H, H,
      d2, D2, nullptr, 0,
      cat2, (long long)D2 * 2 * H);

  // 7) h2[r] = cat2[r] @ w2                    (M=512,N=128,K=256)
  gemm_bf16_wmma<0><<<dim3(D2 / 128, H / BN, R), blk, 0, stream>>>(
      cat2, (long long)D2 * 2 * H, 2 * H,
      w2, 0, H,
      nullptr, 0, nullptr, 0,
      h2, (long long)D2 * H, H,
      2 * H);

  // 8) x2[r] = h2[r] * softmax(attention)
  attention_softmax<<<dim3(R), blk, 0, stream>>>(
      h2, (long long)D2 * H, relv, attv, x2, (long long)D2 * H, D2, H);

  // 9) out = softmax(l2norm(sum_r x2[r]) @ cw)
  finalize<<<dim3(D2), dim3(128), 0, stream>>>(
      x2, (long long)D2 * H, cw, out, H, Cd);
}
